// GPTAttention_27633819582467
// MI455X (gfx1250) — matrix operand order repaired and checked by reading, not compiled, not run on hardware
//
#include <hip/hip_runtime.h>

typedef __attribute__((ext_vector_type(4)))  __bf16 bf16x4;
typedef __attribute__((ext_vector_type(8)))  __bf16 bf16x8;
typedef __attribute__((ext_vector_type(16))) __bf16 bf16x16;
typedef __attribute__((ext_vector_type(4)))  float  f32x4;
typedef __attribute__((ext_vector_type(8)))  float  f32x8;

#define D_MODEL 1024
#define N3      3072
#define N_HEADS 16
#define HEAD_DIM 64
#define BATCH   2
#define SEQ     2048
#define ROWS    (BATCH * SEQ)   // 4096

__device__ __forceinline__ bf16x16 join8(bf16x8 lo, bf16x8 hi) {
  return __builtin_shufflevector(lo, hi, 0, 1, 2, 3, 4, 5, 6, 7,
                                 8, 9, 10, 11, 12, 13, 14, 15);
}
// A-fragment (16x32 bf16): element e <-> K = (e&7) + 16*(e>>3) + 8*laneHi.
// base points at K = 8*laneHi; groups at +0 and +16 are each contiguous.
__device__ __forceinline__ bf16x16 fragA(const __bf16* base) {
  return join8(*(const bf16x8*)base, *(const bf16x8*)(base + 16));
}
// B-fragment (32x16 bf16): element e <-> K = (e&7) + 16*(e>>3) + 8*laneHi (N fixed).
// base points at K = 8*laneHi in a K-contiguous ([N][K]) tile.
__device__ __forceinline__ bf16x16 fragB(const __bf16* base) {
  return join8(*(const bf16x8*)base, *(const bf16x8*)(base + 16));
}
__device__ __forceinline__ bf16x4 cvt4(f32x4 v) {
  bf16x4 r;
  r[0] = (__bf16)v[0]; r[1] = (__bf16)v[1];
  r[2] = (__bf16)v[2]; r[3] = (__bf16)v[3];
  return r;
}
__device__ __forceinline__ float rowMax16(float v) {
#pragma unroll
  for (int m = 1; m < 16; m <<= 1) v = fmaxf(v, __shfl_xor(v, m, 32));
  return v;
}
__device__ __forceinline__ float rowSum16(float v) {
#pragma unroll
  for (int m = 1; m < 16; m <<= 1) v += __shfl_xor(v, m, 32);
  return v;
}

// ---------------------------------------------------------------------------
// Kernel 1: qkv = x @ Wqkv + bqkv -> bf16 Q/K/V buffers [B,H,S,64]
// Block = 128 threads (4 waves); tile 128(M) x 64(N); K-step = 32.
// Per wave: 32x64 slab = 2 A-frags x 4 B-frags = 8 WMMAs per K-step.
// ---------------------------------------------------------------------------
__global__ __launch_bounds__(128)
void GPTAttention_qkv(const float* __restrict__ X, const float* __restrict__ W,
                      const float* __restrict__ bias,
                      __bf16* __restrict__ qb, __bf16* __restrict__ kb,
                      __bf16* __restrict__ vb) {
  __shared__ __align__(16) __bf16 As[128 * 32];  // [row][k]
  __shared__ __align__(16) __bf16 Bs[64 * 32];   // [col][k] (transposed)
  const int tn = blockIdx.x;      // 48 tiles over N3
  const int tm = blockIdx.y;      // 32 tiles over ROWS
  const int tid = threadIdx.x;
  const int wave = tid >> 5, lane = tid & 31;
  const int laneHi = lane >> 4, lane15 = lane & 15;

  f32x8 c[2][4] = {};

  for (int k0 = 0; k0 < D_MODEL; k0 += 32) {
    // stage A: 128x32 fp32 -> bf16 (vector loads, packed 8B stores)
#pragma unroll
    for (int t = 0; t < 8; ++t) {
      const int idx = tid + t * 128;                 // 0..1023 float4 groups
      const int r = idx >> 3, k4 = (idx & 7) << 2;
      f32x4 va = *(const f32x4*)&X[(size_t)(tm * 128 + r) * D_MODEL + k0 + k4];
      *(bf16x4*)&As[r * 32 + k4] = cvt4(va);
    }
    // stage B: 32x64 fp32 -> bf16 transposed into [col][k]
#pragma unroll
    for (int t = 0; t < 4; ++t) {
      const int idx = tid + t * 128;                 // 0..511 float4 groups
      const int kr = idx >> 4, c4 = (idx & 15) << 2;
      f32x4 vw = *(const f32x4*)&W[(size_t)(k0 + kr) * N3 + tn * 64 + c4];
#pragma unroll
      for (int u = 0; u < 4; ++u) Bs[(c4 + u) * 32 + kr] = (__bf16)vw[u];
    }
    if (k0 + 32 < D_MODEL)
      __builtin_prefetch(&X[(size_t)(tm * 128 + (tid >> 1)) * D_MODEL + k0 + 32], 0, 1);
    __syncthreads();

    bf16x16 a[2];
#pragma unroll
    for (int mi = 0; mi < 2; ++mi)
      a[mi] = fragA(&As[(wave * 32 + mi * 16 + lane15) * 32 + (laneHi << 3)]);
#pragma unroll
    for (int ns = 0; ns < 4; ++ns) {
      const bf16x16 b = fragB(&Bs[(ns * 16 + lane15) * 32 + (laneHi << 3)]);
#pragma unroll
      for (int mi = 0; mi < 2; ++mi)
        c[mi][ns] = __builtin_amdgcn_wmma_f32_16x16x32_bf16(
            false, a[mi], false, b, (short)0, c[mi][ns], false, false);
    }
    __syncthreads();
  }

#pragma unroll
  for (int ns = 0; ns < 4; ++ns) {
    const int col = tn * 64 + ns * 16 + lane15;     // 0..3071
    const float bv = bias[col];
    const int which = col >> 10;                    // 0=q 1=k 2=v
    const int d = col & 1023;
    const int h = d >> 6, hdI = d & 63;
    __bf16* dst = (which == 0) ? qb : (which == 1) ? kb : vb;
#pragma unroll
    for (int mi = 0; mi < 2; ++mi)
#pragma unroll
      for (int i = 0; i < 8; ++i) {
        const int row = tm * 128 + wave * 32 + mi * 16 + i + (laneHi << 3);
        const int bb = row >> 11, ss = row & 2047;
        dst[(size_t)((bb * N_HEADS + h) * SEQ + ss) * HEAD_DIM + hdI] =
            (__bf16)(c[mi][ns][i] + bv);
      }
  }
}

// ---------------------------------------------------------------------------
// Kernel 2: causal flash attention per (b,h). Block = 4 waves = 64 query rows.
// Ks kept in natural [kcol][hd] layout (hd contraction contiguous -> b128 frags);
// V staged transposed [hd][krow] so PV B-frags are contiguous too.
// ---------------------------------------------------------------------------
__global__ __launch_bounds__(128)
void GPTAttention_flash(const __bf16* __restrict__ qb,
                        const __bf16* __restrict__ kb,
                        const __bf16* __restrict__ vb,
                        __bf16* __restrict__ ab) {
  __shared__ __align__(16) __bf16 Qs[64 * 64];     // [qrow][hd]
  __shared__ __align__(16) __bf16 Ks[64 * 64];     // [kcol][hd]
  __shared__ __align__(16) __bf16 Vt[64 * 64];     // [hd][krow]
  __shared__ __align__(16) __bf16 Ps[4][16 * 64];  // per-wave P patch [row][kcol]

  const int qt = blockIdx.x;           // 32 query tiles
  const int bh = blockIdx.y;           // 32 (b,h)
  const int bB = bh >> 4, hH = bh & 15;
  const int tid = threadIdx.x;
  const int wave = tid >> 5, lane = tid & 31;
  const int laneHi = lane >> 4, lane15 = lane & 15;

  const __bf16* Qg = qb + (size_t)bh * SEQ * HEAD_DIM;
  const __bf16* Kg = kb + (size_t)bh * SEQ * HEAD_DIM;
  const __bf16* Vg = vb + (size_t)bh * SEQ * HEAD_DIM;

#pragma unroll
  for (int t = 0; t < 4; ++t) {
    const int idx = tid + t * 128;                 // 0..511 8-elem groups
    const int r = idx >> 3, c8 = (idx & 7) << 3;
    *(bf16x8*)&Qs[r * 64 + c8] =
        *(const bf16x8*)&Qg[(size_t)(qt * 64 + r) * HEAD_DIM + c8];
  }
  __syncthreads();

  bf16x16 qa[2];
  {
    const int qrow = wave * 16 + lane15;
#pragma unroll
    for (int kc = 0; kc < 2; ++kc)
      qa[kc] = fragA(&Qs[qrow * 64 + kc * 32 + (laneHi << 3)]);
  }

  f32x8 acc[4] = {};
  float m_i[8], l_i[8];
#pragma unroll
  for (int i = 0; i < 8; ++i) { m_i[i] = -1e30f; l_i[i] = 0.f; }

  for (int j = 0; j <= qt; ++j) {
#pragma unroll
    for (int t = 0; t < 4; ++t) {
      const int idx = tid + t * 128;
      const int r = idx >> 3, c8 = (idx & 7) << 3;
      // K tile: direct vector copy, stays [kcol][hd]
      *(bf16x8*)&Ks[r * 64 + c8] =
          *(const bf16x8*)&Kg[(size_t)(j * 64 + r) * HEAD_DIM + c8];
      // V tile: transpose into [hd][krow]
      bf16x8 vv = *(const bf16x8*)&Vg[(size_t)(j * 64 + r) * HEAD_DIM + c8];
#pragma unroll
      for (int u = 0; u < 8; ++u) Vt[(c8 + u) * 64 + r] = vv[u];
    }
    __syncthreads();

    // scores: 16(M) x 64(N) per wave, K = hd = 64
    f32x8 s[4] = {};
#pragma unroll
    for (int ns = 0; ns < 4; ++ns)
#pragma unroll
      for (int kc = 0; kc < 2; ++kc) {
        const bf16x16 bk =
            fragB(&Ks[(ns * 16 + lane15) * 64 + kc * 32 + (laneHi << 3)]);
        s[ns] = __builtin_amdgcn_wmma_f32_16x16x32_bf16(false, qa[kc], false, bk,
                                                        (short)0, s[ns], false, false);
      }

    const bool diag = (j == qt);
#pragma unroll
    for (int ns = 0; ns < 4; ++ns)
#pragma unroll
      for (int i = 0; i < 8; ++i) {
        float v = s[ns][i] * 0.125f;                 // 1/sqrt(64)
        if (diag) {
          const int kcol = j * 64 + ns * 16 + lane15;
          const int qr   = qt * 64 + wave * 16 + i + (laneHi << 3);
          if (kcol > qr) v = -1e30f;
        }
        s[ns][i] = v;
      }

    // online softmax
#pragma unroll
    for (int i = 0; i < 8; ++i) {
      float mx = fmaxf(fmaxf(s[0][i], s[1][i]), fmaxf(s[2][i], s[3][i]));
      mx = rowMax16(mx);
      const float nm = fmaxf(m_i[i], mx);
      const float scale = __expf(m_i[i] - nm);
      m_i[i] = nm;
      l_i[i] *= scale;
#pragma unroll
      for (int ns = 0; ns < 4; ++ns) acc[ns][i] *= scale;
      float rs = 0.f;
#pragma unroll
      for (int ns = 0; ns < 4; ++ns) {
        const float p = __expf(s[ns][i] - nm);
        s[ns][i] = p;
        rs += p;
      }
      l_i[i] += rowSum16(rs);
    }

    // C-layout -> A-layout via wave-private LDS patch
#pragma unroll
    for (int ns = 0; ns < 4; ++ns)
#pragma unroll
      for (int i = 0; i < 8; ++i)
        Ps[wave][(i + (laneHi << 3)) * 64 + ns * 16 + lane15] = (__bf16)s[ns][i];

    // acc += P @ V
#pragma unroll
    for (int kc = 0; kc < 2; ++kc) {
      const bf16x16 pa =
          fragA(&Ps[wave][lane15 * 64 + kc * 32 + (laneHi << 3)]);
#pragma unroll
      for (int nc = 0; nc < 4; ++nc) {
        const bf16x16 bv =
            fragB(&Vt[(nc * 16 + lane15) * 64 + kc * 32 + (laneHi << 3)]);
        acc[nc] = __builtin_amdgcn_wmma_f32_16x16x32_bf16(false, pa, false, bv,
                                                          (short)0, acc[nc], false, false);
      }
    }
    __syncthreads();
  }

#pragma unroll
  for (int nc = 0; nc < 4; ++nc) {
    const int hdc = nc * 16 + lane15;
#pragma unroll
    for (int i = 0; i < 8; ++i) {
      const int srow = qt * 64 + wave * 16 + i + (laneHi << 3);
      const float o = acc[nc][i] / l_i[i];
      ab[(size_t)(bB * SEQ + srow) * D_MODEL + hH * HEAD_DIM + hdc] = (__bf16)o;
    }
  }
}

// ---------------------------------------------------------------------------
// Kernel 3: out = attn @ Wo + bo (A already bf16; W converted while staging)
// Same 128x64 tiling as kernel 1.
// ---------------------------------------------------------------------------
__global__ __launch_bounds__(128)
void GPTAttention_oproj(const __bf16* __restrict__ A, const float* __restrict__ W,
                        const float* __restrict__ bias, float* __restrict__ out) {
  __shared__ __align__(16) __bf16 As[128 * 32];  // [row][k]
  __shared__ __align__(16) __bf16 Bs[64 * 32];   // [col][k]
  const int tn = blockIdx.x;      // 16 tiles over D_MODEL
  const int tm = blockIdx.y;      // 32 tiles over ROWS
  const int tid = threadIdx.x;
  const int wave = tid >> 5, lane = tid & 31;
  const int laneHi = lane >> 4, lane15 = lane & 15;

  f32x8 c[2][4] = {};

  for (int k0 = 0; k0 < D_MODEL; k0 += 32) {
#pragma unroll
    for (int t = 0; t < 4; ++t) {                       // A: b128 bf16 copies
      const int idx = tid + t * 128;                    // 0..511 8-elem groups
      const int r = idx >> 2, k8 = (idx & 3) << 3;
      *(bf16x8*)&As[r * 32 + k8] =
          *(const bf16x8*)&A[(size_t)(tm * 128 + r) * D_MODEL + k0 + k8];
    }
#pragma unroll
    for (int t = 0; t < 4; ++t) {                       // B: fp32 -> bf16 transpose
      const int idx = tid + t * 128;
      const int kr = idx >> 4, c4 = (idx & 15) << 2;
      f32x4 vw = *(const f32x4*)&W[(size_t)(k0 + kr) * D_MODEL + tn * 64 + c4];
#pragma unroll
      for (int u = 0; u < 4; ++u) Bs[(c4 + u) * 32 + kr] = (__bf16)vw[u];
    }
    __syncthreads();

    bf16x16 a[2];
#pragma unroll
    for (int mi = 0; mi < 2; ++mi)
      a[mi] = fragA(&As[(wave * 32 + mi * 16 + lane15) * 32 + (laneHi << 3)]);
#pragma unroll
    for (int ns = 0; ns < 4; ++ns) {
      const bf16x16 b = fragB(&Bs[(ns * 16 + lane15) * 32 + (laneHi << 3)]);
#pragma unroll
      for (int mi = 0; mi < 2; ++mi)
        c[mi][ns] = __builtin_amdgcn_wmma_f32_16x16x32_bf16(
            false, a[mi], false, b, (short)0, c[mi][ns], false, false);
    }
    __syncthreads();
  }

#pragma unroll
  for (int ns = 0; ns < 4; ++ns) {
    const int col = tn * 64 + ns * 16 + lane15;
    const float bv = bias[col];
#pragma unroll
    for (int mi = 0; mi < 2; ++mi)
#pragma unroll
      for (int i = 0; i < 8; ++i) {
        const int row = tm * 128 + wave * 32 + mi * 16 + i + (laneHi << 3);
        out[(size_t)row * D_MODEL + col] = c[mi][ns][i] + bv;
      }
  }
}

// ---------------------------------------------------------------------------
extern "C" void kernel_launch(void* const* d_in, const int* in_sizes, int n_in,
                              void* d_out, int out_size, void* d_ws, size_t ws_size,
                              hipStream_t stream) {
  (void)in_sizes; (void)n_in; (void)out_size; (void)ws_size;
  const float* x    = (const float*)d_in[0];
  // d_in[1]: causal mask (bool) — causality hardcoded in the flash kernel
  const float* Wqkv = (const float*)d_in[2];
  const float* bqkv = (const float*)d_in[3];
  const float* Wo   = (const float*)d_in[4];
  const float* bo   = (const float*)d_in[5];

  char* ws = (char*)d_ws;
  const size_t HBUF_BYTES = (size_t)BATCH * N_HEADS * SEQ * HEAD_DIM * sizeof(__bf16); // 8 MB
  __bf16* qb = (__bf16*)(ws + 0 * HBUF_BYTES);
  __bf16* kb = (__bf16*)(ws + 1 * HBUF_BYTES);
  __bf16* vb = (__bf16*)(ws + 2 * HBUF_BYTES);
  __bf16* ab = (__bf16*)(ws + 3 * HBUF_BYTES);

  dim3 blk(128);
  GPTAttention_qkv  <<<dim3(N3 / 64, ROWS / 128), blk, 0, stream>>>(x, Wqkv, bqkv, qb, kb, vb);
  GPTAttention_flash<<<dim3(SEQ / 64, BATCH * N_HEADS), blk, 0, stream>>>(qb, kb, vb, ab);
  GPTAttention_oproj<<<dim3(D_MODEL / 64, ROWS / 128), blk, 0, stream>>>(ab, Wo, bo, (float*)d_out);
}
